// SparseAttention_AFM_90005334655072
// MI455X (gfx1250) — compile-verified, matrix operand order checked by reading, not run on hardware
//
#include <hip/hip_runtime.h>
#include <hip/hip_bf16.h>

// SparseAttention_AFM on MI455X (gfx1250), bf16 WMMA pipeline.
// Operands are pre-packed into per-lane WMMA fragment order; workgroup-shared
// operands are staged into LDS via the CDNA5 async paths:
//   - QKV weights: per-lane GLOBAL_LOAD_ASYNC_TO_LDS_B128 (+ s_wait_asynccnt)
//   - attention B panels: Tensor Data Mover, double-buffered
//     (tensor_load_to_lds + s_wait_tensorcnt)

typedef __bf16 bf16;
typedef __attribute__((ext_vector_type(16))) __bf16 bf16x16;  // 32B: one wmma A/B fragment per lane
typedef __attribute__((ext_vector_type(8)))  float  v8f;      // one wmma C/D fragment per lane
typedef __attribute__((ext_vector_type(4)))  unsigned u32x4;
typedef __attribute__((ext_vector_type(8)))  unsigned u32x8;

#define WMMA_BF16(a, b, c)                                                     \
  __builtin_amdgcn_wmma_f32_16x16x32_bf16(false, (a), false, (b), (short)0,    \
                                          (c), false, false)

constexpr int Bn   = 8;
constexpr int Nn   = 2048;
constexpr int Dn   = 512;
constexpr int TwoD = 1024;
constexpr int BN   = Bn * Nn;  // 16384 rows for the QKV GEMM

// ---- CDNA5 async global->LDS copy (16B per lane), GV addressing ------------
__device__ __forceinline__ void async_copy_b128(unsigned lds_byte_addr,
                                                const void* gsrc) {
  asm volatile("global_load_async_to_lds_b128 %0, %1, off"
               :: "v"(lds_byte_addr), "v"((unsigned long long)gsrc)
               : "memory");
}
__device__ __forceinline__ void wait_asynccnt0() {
  asm volatile("s_wait_asynccnt 0x0" ::: "memory");
}

// ---- CDNA5 Tensor Data Mover: 1-D contiguous DMA, global -> LDS ------------
// D# group0: count=1 (valid), lds_addr[63:32], global_addr[120:64], type=2.
// D# group1: data_size=3 (8B), tensor_dim0 = tile_dim0 = n8 elements,
//            tensor_dim1 = tile_dim1 = 1, tensor_dim0_stride = n8.
__device__ __forceinline__ void tdm_load_1d(unsigned lds_byte_addr,
                                            const void* gsrc, unsigned n8) {
  unsigned long long ga = (unsigned long long)gsrc;
  u32x4 g0;
  g0[0] = 1u;                                                   // count=1
  g0[1] = lds_byte_addr;                                        // lds_addr
  g0[2] = (unsigned)ga;                                         // global_addr lo
  g0[3] = (unsigned)((ga >> 32) & 0x01FFFFFFu) | (2u << 30);    // addr hi | type=2
  u32x8 g1;
  g1[0] = 3u << 16;                          // workgroup_mask=0, data_size=8B
  g1[1] = (n8 & 0xFFFFu) << 16;              // tensor_dim0[15:0] @ bits 63:48
  g1[2] = (n8 >> 16) | (1u << 16);           // tensor_dim0 hi | tensor_dim1=1
  g1[3] = (n8 & 0xFFFFu) << 16;              // tile_dim0 @ bits 127:112
  g1[4] = 1u;                                // tile_dim1=1, tile_dim2=0
  g1[5] = n8;                                // tensor_dim0_stride lo
  g1[6] = 0u;
  g1[7] = 0u;
  asm volatile("tensor_load_to_lds %0, %1" :: "s"(g0), "s"(g1) : "memory");
}
__device__ __forceinline__ void wait_tensorcnt0() {
  asm volatile("s_wait_tensorcnt 0x0" ::: "memory");
}
__device__ __forceinline__ void wait_tensorcnt4() {
  // TDM ops from one wave complete in order: cnt<=4 means the previous
  // chunk's 4 DMAs are done while the next chunk's 4 remain in flight.
  asm volatile("s_wait_tensorcnt 0x4" ::: "memory");
}

// ---------------------------------------------------------------------------
// Fragment packing. ISA 16-bit A-layout (16x32 MxK): lane l (m = l&15,
// half = l>>4) holds K = {half*8 .. half*8+7} then {16+half*8 ..}. The bf16
// B-layout (32x16 KxN) mirrors it with N on lanes; for y = x @ W^T the B
// fragment of W is exactly the A-fragment pack of row-major W.
// ---------------------------------------------------------------------------
__global__ void pack_a_bf16_kernel(const float* __restrict__ src,
                                   bf16* __restrict__ dst, int R, int C) {
  int tid = blockIdx.x * blockDim.x + threadIdx.x;
  int ktiles = C >> 5;
  int total = (R >> 4) * ktiles * 32;
  if (tid >= total) return;
  int lane = tid & 31;
  int tile = tid >> 5;  // mt*ktiles + kt
  int kt = tile % ktiles;
  int mt = tile / ktiles;
  int m = lane & 15;
  int half = lane >> 4;
  const float* rp = src + (size_t)(mt * 16 + m) * C + (size_t)kt * 32;
  bf16x16 v;
#pragma unroll
  for (int j = 0; j < 8; ++j) v[j] = (bf16)rp[half * 8 + j];
#pragma unroll
  for (int j = 0; j < 8; ++j) v[8 + j] = (bf16)rp[16 + half * 8 + j];
  ((bf16x16*)dst)[(size_t)tile * 32 + lane] = v;
}

// dis_weight^T, exp(), and A-fragment packing fused in one pass.
__global__ void dis_pack_kernel(const float* __restrict__ dis,
                                const float* __restrict__ alpha,
                                bf16* __restrict__ dst) {
  int tid = blockIdx.x * blockDim.x + threadIdx.x;  // B*128*64*32 threads
  int lane = tid & 31;
  int tile = tid >> 5;  // b*(128*64) + mt*64 + kt
  int kt = tile & 63;
  int mtb = tile >> 6;
  int mt = mtb & 127;
  int b = mtb >> 7;
  float coef = alpha[0] * 11.0f;  // log2(2048) = 11
  int j = mt * 16 + (lane & 15);
  int half = lane >> 4;
  const float* base = dis + (size_t)b * Nn * Nn;
  bf16x16 v;
#pragma unroll
  for (int t = 0; t < 8; ++t) {
    int i = kt * 32 + half * 8 + t;
    v[t] = (bf16)__expf(-coef * base[(size_t)i * Nn + j]);
  }
#pragma unroll
  for (int t = 0; t < 8; ++t) {
    int i = kt * 32 + 16 + half * 8 + t;
    v[8 + t] = (bf16)__expf(-coef * base[(size_t)i * Nn + j]);
  }
  ((bf16x16*)dst)[(size_t)tile * 32 + lane] = v;
}

// ---------------------------------------------------------------------------
// GEMM 1: fused QKV, workgroup-cooperative. Block = 8 waves = 8 row tiles
// sharing one column tile; the full K-panels of Wq/Wk/Wv (3 x 16KB,
// contiguous in fragment order) are async-staged into LDS once.
// ---------------------------------------------------------------------------
__global__ void qkv_wmma_kernel(const bf16* __restrict__ xp,
                                const bf16* __restrict__ wqp,
                                const bf16* __restrict__ wkp,
                                const bf16* __restrict__ wvp,
                                float* __restrict__ sigq,
                                bf16* __restrict__ mmat) {
  __shared__ __align__(32) unsigned char smem[49152];  // 3 x 16KB weight panels
  int t = threadIdx.x;
  int lane = t & 31;
  int wave = t >> 5;
  int ct = blockIdx.x & 31;               // column tile over D/16
  int rt = (blockIdx.x >> 5) * 8 + wave;  // row tile over BN/16

  unsigned lds_base = (unsigned)(unsigned long long)(void*)&smem[0];
  // Stage the 3 weight K-panels: 48KB / 256 threads = 12 x b128 per thread.
  {
    const char* wsrc0 = (const char*)wqp + (size_t)ct * 16384 + (size_t)t * 64;
    const char* wsrc1 = (const char*)wkp + (size_t)ct * 16384 + (size_t)t * 64;
    const char* wsrc2 = (const char*)wvp + (size_t)ct * 16384 + (size_t)t * 64;
#pragma unroll
    for (int s = 0; s < 4; ++s) {
      async_copy_b128(lds_base + 0u     + t * 64 + s * 16, wsrc0 + s * 16);
      async_copy_b128(lds_base + 16384u + t * 64 + s * 16, wsrc1 + s * 16);
      async_copy_b128(lds_base + 32768u + t * 64 + s * 16, wsrc2 + s * 16);
    }
    wait_asynccnt0();
    __syncthreads();
  }

  const bf16x16* A = (const bf16x16*)xp;
  v8f aq = {}, ak = {}, av = {};
#pragma unroll 4
  for (int kt = 0; kt < 16; ++kt) {  // K = 512 / 32
    bf16x16 a  = A[(size_t)(rt * 16 + kt) * 32 + lane];
    bf16x16 bq = *(const bf16x16*)(smem + 0     + kt * 1024 + lane * 32);
    bf16x16 bk = *(const bf16x16*)(smem + 16384 + kt * 1024 + lane * 32);
    bf16x16 bv = *(const bf16x16*)(smem + 32768 + kt * 1024 + lane * 32);
    if (kt + 1 < 16)
      __builtin_prefetch(&A[(size_t)(rt * 16 + kt + 1) * 32 + lane], 0, 1);
    aq = WMMA_BF16(a, bq, aq);
    ak = WMMA_BF16(a, bk, ak);
    av = WMMA_BF16(a, bv, av);
  }
  // C layout: VGPR r -> row m0+r, lanes 0-15 m0=0, lanes 16-31 m0=8.
  int m0 = (lane >> 4) * 8;
  int col = ct * 16 + (lane & 15);
#pragma unroll
  for (int r = 0; r < 8; ++r) {
    int row = rt * 16 + m0 + r;
    sigq[(size_t)row * Dn + col] = 1.0f / (1.0f + __expf(-aq[r]));
    float kw = __expf(ak[r]);
    mmat[(size_t)row * TwoD + col]      = (bf16)(kw * av[r]);
    mmat[(size_t)row * TwoD + Dn + col] = (bf16)kw;
  }
}

// Repack row-major mmat [N x 2D] per batch into WMMA B-fragment order.
__global__ void mpack_kernel(const bf16* __restrict__ mmat,
                             bf16* __restrict__ mp) {
  int tid = blockIdx.x * blockDim.x + threadIdx.x;  // B*64*64*32 threads
  int lane = tid & 31;
  int tile = tid >> 5;  // b*(64*64) + nt*64 + kt
  int kt = tile & 63;
  int ntb = tile >> 6;
  int nt = ntb & 63;
  int b = ntb >> 6;
  int col = nt * 16 + (lane & 15);
  int half = lane >> 4;
  const bf16* base = mmat + (size_t)b * Nn * TwoD;
  bf16x16 v;
#pragma unroll
  for (int t = 0; t < 8; ++t)
    v[t] = base[(size_t)(kt * 32 + half * 8 + t) * TwoD + col];
#pragma unroll
  for (int t = 0; t < 8; ++t)
    v[8 + t] = base[(size_t)(kt * 32 + 16 + half * 8 + t) * TwoD + col];
  ((bf16x16*)mp)[(size_t)tile * 32 + lane] = v;
}

// ---------------------------------------------------------------------------
// GEMM 2: per batch, dis^T [2048x2048] @ [kw*v | kw] [2048x1024].
// Block = 8 waves; each wave computes a 2x2 super-tile: row tiles (jt0, jt0+1)
// x column-tile pairs (dtb, dtb+32) and (dtb+16, dtb+48) -> 8 accumulators,
// 8 WMMAs per 2 global A loads + 4 LDS B loads (21 FLOP/byte at the wave).
// The 4 shared B panels are DMA'd by the TDM into double-buffered LDS (2x32KB,
// 8 K-steps per chunk); s_wait_tensorcnt 4 overlaps DMA with compute.
// ---------------------------------------------------------------------------
__device__ __forceinline__ void attn_store_tile(float* __restrict__ out,
                                                const float* __restrict__ sigq,
                                                int b, int jt, int dt, int lane,
                                                v8f w1, v8f w2) {
  int m0 = (lane >> 4) * 8;
  int d = dt * 16 + (lane & 15);
#pragma unroll
  for (int r = 0; r < 8; ++r) {
    int j = jt * 16 + m0 + r;
    size_t idx = ((size_t)b * Nn + j) * Dn + d;
    out[idx] = sigq[idx] * w1[r] / w2[r];
  }
}

__device__ __forceinline__ void attn_stage_chunk(unsigned lds,
                                                 const char* BmB, int dtb,
                                                 int chunk) {
  int k0 = chunk * 8;  // 8 K-step tiles, 8KB contiguous per panel
  tdm_load_1d(lds + 0u,     BmB + (size_t)((dtb)      * 64 + k0) * 1024, 1024);
  tdm_load_1d(lds + 8192u,  BmB + (size_t)((dtb + 16) * 64 + k0) * 1024, 1024);
  tdm_load_1d(lds + 16384u, BmB + (size_t)((dtb + 32) * 64 + k0) * 1024, 1024);
  tdm_load_1d(lds + 24576u, BmB + (size_t)((dtb + 48) * 64 + k0) * 1024, 1024);
}

__global__ void attn_wmma_kernel(const bf16* __restrict__ dp,
                                 const bf16* __restrict__ mp,
                                 const float* __restrict__ sigq,
                                 float* __restrict__ out) {
  __shared__ __align__(32) unsigned char smem[65536];  // 2 bufs x 4 x 8KB panels
  int t = threadIdx.x;
  int lane = t & 31;
  int wave = t >> 5;
  int dtb = blockIdx.x & 15;             // column pair block
  int rem = blockIdx.x >> 4;             // 0..63
  int jt0 = (rem & 7) * 16 + wave * 2;   // two row tiles: jt0, jt0+1
  int b = rem >> 3;                      // batch

  unsigned lds_base = (unsigned)(unsigned long long)(void*)&smem[0];
  const bf16x16* A = (const bf16x16*)dp + (size_t)b * (128 * 64) * 32;
  const char* BmB = (const char*)mp + (size_t)b * (64 * 64) * 1024;  // 1KB/tile

  v8f aX1_0 = {}, aX2_0 = {}, aY1_0 = {}, aY2_0 = {};
  v8f aX1_1 = {}, aX2_1 = {}, aY1_1 = {}, aY2_1 = {};

  if (t < 32) attn_stage_chunk(lds_base, BmB, dtb, 0);  // wave 0 drives TDM

  for (int chunk = 0; chunk < 8; ++chunk) {  // 8 x 8 K-steps = K 2048
    int buf = chunk & 1;
    if (t < 32) {
      if (chunk + 1 < 8)
        attn_stage_chunk(lds_base + (unsigned)(buf ^ 1) * 32768u, BmB, dtb,
                         chunk + 1);
      wait_tensorcnt4();  // current chunk landed; next chunk stays in flight
    }
    __syncthreads();
    const unsigned char* sb = smem + buf * 32768;
#pragma unroll
    for (int c2 = 0; c2 < 8; ++c2) {
      int kt = chunk * 8 + c2;
      bf16x16 a0 = A[(size_t)(jt0 * 64 + kt) * 32 + lane];
      bf16x16 a1 = A[(size_t)((jt0 + 1) * 64 + kt) * 32 + lane];
      bf16x16 bX1 = *(const bf16x16*)(sb + 0     + c2 * 1024 + lane * 32);
      bf16x16 bY1 = *(const bf16x16*)(sb + 8192  + c2 * 1024 + lane * 32);
      bf16x16 bX2 = *(const bf16x16*)(sb + 16384 + c2 * 1024 + lane * 32);
      bf16x16 bY2 = *(const bf16x16*)(sb + 24576 + c2 * 1024 + lane * 32);
      if (kt + 1 < 64) {
        __builtin_prefetch(&A[(size_t)(jt0 * 64 + kt + 1) * 32 + lane], 0, 1);
        __builtin_prefetch(&A[(size_t)((jt0 + 1) * 64 + kt + 1) * 32 + lane], 0, 1);
      }
      aX1_0 = WMMA_BF16(a0, bX1, aX1_0);
      aX1_1 = WMMA_BF16(a1, bX1, aX1_1);
      aY1_0 = WMMA_BF16(a0, bY1, aY1_0);
      aY1_1 = WMMA_BF16(a1, bY1, aY1_1);
      aX2_0 = WMMA_BF16(a0, bX2, aX2_0);
      aX2_1 = WMMA_BF16(a1, bX2, aX2_1);
      aY2_0 = WMMA_BF16(a0, bY2, aY2_0);
      aY2_1 = WMMA_BF16(a1, bY2, aY2_1);
    }
    __syncthreads();  // buf fully consumed before it is re-staged
  }
  if (t < 32) wait_tensorcnt0();  // quiesce TDM before wave exit

  attn_store_tile(out, sigq, b, jt0,     dtb,      lane, aX1_0, aX2_0);
  attn_store_tile(out, sigq, b, jt0,     dtb + 16, lane, aY1_0, aY2_0);
  attn_store_tile(out, sigq, b, jt0 + 1, dtb,      lane, aX1_1, aX2_1);
  attn_store_tile(out, sigq, b, jt0 + 1, dtb + 16, lane, aY1_1, aY2_1);
}

extern "C" void kernel_launch(void* const* d_in, const int* in_sizes, int n_in,
                              void* d_out, int out_size, void* d_ws,
                              size_t ws_size, hipStream_t stream) {
  (void)in_sizes; (void)n_in; (void)out_size; (void)ws_size;
  const float* x     = (const float*)d_in[0];
  const float* dis   = (const float*)d_in[1];
  const float* Wq    = (const float*)d_in[2];
  const float* Wk    = (const float*)d_in[3];
  const float* Wv    = (const float*)d_in[4];
  const float* alpha = (const float*)d_in[5];
  float* out = (float*)d_out;

  // Workspace carve-up (~178 MiB total), all 256B-aligned offsets.
  char* ws = (char*)d_ws;
  size_t off = 0;
  bf16*  xp   = (bf16*)(ws + off); off += (size_t)BN * Dn * 2;        // 16 MiB
  bf16*  wqp  = (bf16*)(ws + off); off += (size_t)Dn * Dn * 2;
  bf16*  wkp  = (bf16*)(ws + off); off += (size_t)Dn * Dn * 2;
  bf16*  wvp  = (bf16*)(ws + off); off += (size_t)Dn * Dn * 2;
  float* sigq = (float*)(ws + off); off += (size_t)BN * Dn * 4;       // 32 MiB
  bf16*  mmat = (bf16*)(ws + off); off += (size_t)BN * TwoD * 2;      // 32 MiB
  bf16*  mp   = (bf16*)(ws + off); off += (size_t)BN * TwoD * 2;      // 32 MiB
  bf16*  dp   = (bf16*)(ws + off); off += (size_t)Bn * Nn * Nn * 2;   // 64 MiB

  const int TB = 256;
  // 1) pack x and the three weight matrices to bf16 fragments
  pack_a_bf16_kernel<<<((BN / 16) * (Dn / 32) * 32) / TB, TB, 0, stream>>>(x, xp, BN, Dn);
  pack_a_bf16_kernel<<<((Dn / 16) * (Dn / 32) * 32) / TB, TB, 0, stream>>>(Wq, wqp, Dn, Dn);
  pack_a_bf16_kernel<<<((Dn / 16) * (Dn / 32) * 32) / TB, TB, 0, stream>>>(Wk, wkp, Dn, Dn);
  pack_a_bf16_kernel<<<((Dn / 16) * (Dn / 32) * 32) / TB, TB, 0, stream>>>(Wv, wvp, Dn, Dn);
  // 2) dis_weight^T + exp + pack (reads alpha on device: graph-capture safe)
  dis_pack_kernel<<<(Bn * (Nn / 16) * (Nn / 32) * 32) / TB, TB, 0, stream>>>(dis, alpha, dp);
  // 3) fused QKV GEMM (WMMA, async-LDS staged weights) + pointwise epilogue
  qkv_wmma_kernel<<<((BN / 16 / 8) * (Dn / 16)), TB, 0, stream>>>(xp, wqp, wkp, wvp, sigq, mmat);
  // 4) repack [kw*v | kw] into B fragments
  mpack_kernel<<<(Bn * (TwoD / 16) * (Nn / 32) * 32) / TB, TB, 0, stream>>>(mmat, mp);
  // 5) aggregation GEMM (WMMA 2x2 super-tiles, double-buffered TDM B panels)
  attn_wmma_kernel<<<(Bn * (Nn / 16 / 16) * (Dn / 16 / 2)), TB, 0, stream>>>(dp, mp, sigq, out);
}
